// EGNN_14242111554069
// MI455X (gfx1250) — matrix operand order repaired and checked by reading, not compiled, not run on hardware
//
#include <hip/hip_runtime.h>
#include <math.h>

// ---------------------------------------------------------------------------
// EGNN layer for MI455X (gfx1250, wave32, WMMA 16x16x32 bf16 / f32 acc).
//
// Pipeline (all on `stream`):
//   0) zero agg/v_sum/cnt
//   1) prep: swizzle all MLP weights into per-lane WMMA B-operand images (bf16)
//   2) embed: h_emb = h @ emb_W + emb_b  (f32 + bf16 copy; 12.8MB -> L2-resident)
//   3) edge kernel (1 wave = 16 edges): gather h[row],h[col] as bf16 WMMA A ops,
//      edge MLP (16 wmma) -> relu -> LDS -> layer2 (8 wmma) -> atomic scatter agg,
//      w-MLP (8 wmma + dot) -> atomic scatter v_sum/cnt
//   4) node kernel (1 wave = 16 nodes): [h_emb|agg] @ nW1 (16 wmma) -> relu ->
//      LDS -> @ nW2 (8 wmma) -> h_out = h_emb + ...
//   5) v finalize + x copy
// ---------------------------------------------------------------------------

typedef __attribute__((ext_vector_type(16))) __bf16         v16bf;
typedef __attribute__((ext_vector_type(8)))  float          v8f;
typedef __attribute__((ext_vector_type(8)))  unsigned short u16x8;
typedef __attribute__((ext_vector_type(16))) unsigned short u16x16;

#define WMMA_BF16(A, B, C) \
  __builtin_amdgcn_wmma_f32_16x16x32_bf16(false, (A), false, (B), (short)0, (C), false, false)

__device__ __forceinline__ unsigned short f2bf_bits(float f) {
  unsigned u = __builtin_bit_cast(unsigned, f);
  u += 0x7FFFu + ((u >> 16) & 1u);              // round-to-nearest-even
  return (unsigned short)(u >> 16);
}

__device__ __forceinline__ v16bf join16(u16x8 lo, u16x8 hi) {
  u16x16 u = __builtin_shufflevector(lo, hi, 0,1,2,3,4,5,6,7,8,9,10,11,12,13,14,15);
  return __builtin_bit_cast(v16bf, u);
}

// A-operand slice (16-bit A 16x32 layout): lane holds K-runs [koff..koff+7] and
// [koff+16..koff+23] of row m = lane&15; p must point at element koff of that row.
__device__ __forceinline__ v16bf loadA16(const unsigned short* p) {
  u16x8 lo = *(const u16x8*)p;
  u16x8 hi = *(const u16x8*)(p + 16);
  return join16(lo, hi);
}

// A-operand slice built from f32 source (same run structure), converting to bf16.
__device__ __forceinline__ v16bf cvtA16(const float* p) {
  u16x16 u;
#pragma unroll
  for (int i = 0; i < 8; ++i) {
    u[i]     = f2bf_bits(p[i]);
    u[8 + i] = f2bf_bits(p[16 + i]);
  }
  return __builtin_bit_cast(v16bf, u);
}

// Pre-swizzled B-operand image: 32 lanes x 16 bf16 contiguous (1KB per image).
__device__ __forceinline__ v16bf loadB(const unsigned short* wimg, int img, int lane) {
  const unsigned short* p = wimg + img * 512 + lane * 16;
  u16x8 lo = *(const u16x8*)p;
  u16x8 hi = *(const u16x8*)(p + 8);
  return join16(lo, hi);
}

#define WAIT_LDS() asm volatile("s_wait_dscnt 0" ::: "memory")

// ---------------------------------------------------------------------------
// 0) zero scratch
__global__ void egnn_zero(float* __restrict__ p, size_t n) {
  size_t i = (size_t)blockIdx.x * blockDim.x + threadIdx.x;
  if (i < n) p[i] = 0.0f;
}

// ---------------------------------------------------------------------------
// 1) build WMMA B-operand images from f32 weights.
// B layout (32x16 bf16): lane L -> col N = L&15; element e -> K = (e&7) + (e>=8?16:0) + (L>=16?8:0).
// Image table: [0..7] eW1 rows 0..63, [8..15] eW1 rows 64..127, [16..23] eW2,
// [24..31] wW1, [32..47] nW1 (K=128 -> 4 K-halves), [48..55] nW2.
// Within a group: img = kh*4 + nt  (nt = 16-col tile).
__global__ void egnn_prep(const float* __restrict__ eW1, const float* __restrict__ eW2,
                          const float* __restrict__ wW1, const float* __restrict__ nW1,
                          const float* __restrict__ nW2, unsigned short* __restrict__ wimg) {
  int b = blockIdx.x;            // 56 images
  int t = threadIdx.x;           // 512 = 32 lanes x 16 elements
  int L = t >> 4, e = t & 15;
  const float* W; int rowbase = 0, kh, nt;
  if      (b <  8) { W = eW1; rowbase = 0;  kh = b >> 2;        nt = b & 3; }
  else if (b < 16) { W = eW1; rowbase = 64; kh = (b - 8) >> 2;  nt = (b - 8) & 3; }
  else if (b < 24) { W = eW2;               kh = (b - 16) >> 2; nt = (b - 16) & 3; }
  else if (b < 32) { W = wW1;               kh = (b - 24) >> 2; nt = (b - 24) & 3; }
  else if (b < 48) { W = nW1;               kh = (b - 32) >> 2; nt = (b - 32) & 3; }
  else             { W = nW2;               kh = (b - 48) >> 2; nt = (b - 48) & 3; }
  int K  = rowbase + kh * 32 + (e & 7) + ((e >= 8) ? 16 : 0) + ((L >= 16) ? 8 : 0);
  int Nn = nt * 16 + (L & 15);
  wimg[b * 512 + L * 16 + e] = f2bf_bits(W[K * 64 + Nn]);
}

// ---------------------------------------------------------------------------
// 2) node embedding: h_emb = h(Nx16) @ emb_W(16x64) + emb_b ; also bf16 copy.
__global__ void egnn_embed(const float* __restrict__ h, const float* __restrict__ W,
                           const float* __restrict__ b, float* __restrict__ hemb,
                           unsigned short* __restrict__ hbf, int N) {
  int i = blockIdx.x * blockDim.x + threadIdx.x;
  if (i >= N * 64) return;
  int node = i >> 6, j = i & 63;
  const float* hr = h + (size_t)node * 16;
  float a = b[j];
#pragma unroll
  for (int k = 0; k < 16; ++k) a = fmaf(hr[k], W[k * 64 + j], a);
  hemb[i] = a;
  hbf[i]  = f2bf_bits(a);
}

// ---------------------------------------------------------------------------
// 3) fused edge kernel: one wave per 16-edge tile, 8 waves per block.
__global__ void __launch_bounds__(256)
egnn_edge(const int* __restrict__ ei, const float* __restrict__ x,
          const unsigned short* __restrict__ hbf,
          const float* __restrict__ eW1, const float* __restrict__ eb1,
          const float* __restrict__ eb2, const float* __restrict__ wb1,
          const float* __restrict__ wW2, const float* __restrict__ wb2,
          const unsigned short* __restrict__ wimg,
          float* __restrict__ agg, float* __restrict__ vsum, float* __restrict__ cnt,
          int N, int E) {
  __shared__ __align__(16) unsigned short s_hid[8][16 * 64]; // per-wave 16x64 bf16 tile
  __shared__ __align__(16) float          s_hw [8][16 * 64]; // per-wave 16x64 f32 tile

  const int wv   = threadIdx.x >> 5;
  const int lane = threadIdx.x & 31;
  const int tile = blockIdx.x * 8 + wv;
  const int e0   = tile * 16;
  if (e0 >= E) return;                        // uniform per wave -> EXEC stays all-1s

  const int m    = lane & 15;
  const int moff = (lane >= 16) ? 8 : 0;
  const int k0   = moff;                      // A K-run offsets: kh*32 + moff
  const int k1   = 32 + moff;

  int em = e0 + m; if (em >= E) em = E - 1;   // clamp (scatter is guarded)
  const int row = ei[em];
  const int col = ei[E + em];

  // edge geometry (each lane-half computes its own copy; broadcast via shfl)
  const float dx = x[row * 3 + 0] - x[col * 3 + 0];
  const float dy = x[row * 3 + 1] - x[col * 3 + 1];
  const float dz = x[row * 3 + 2] - x[col * 3 + 2];
  const float rd = dx * dx + dy * dy + dz * dz;
  const float th = atan2f(dy, dx);
  const float ph = atan2f(dz, sqrtf(dx * dx + dy * dy));
  const float s0 = rd * cosf(2.0f * th);
  const float s1 = rd * sinf(2.0f * th);
  const float s2 = rd * ph;

  // ---- edge MLP layer 1: hidden = hrow@eW1[0:64] + hcol@eW1[64:128] (+rank1+bias)
  const unsigned short* hr = hbf + (size_t)row * 64;
  const unsigned short* hc = hbf + (size_t)col * 64;
  const v16bf ar0 = loadA16(hr + k0), ar1 = loadA16(hr + k1);
  const v16bf ac0 = loadA16(hc + k0), ac1 = loadA16(hc + k1);

  v8f acc[4];
#pragma unroll
  for (int nt = 0; nt < 4; ++nt) {
    v8f a = {0, 0, 0, 0, 0, 0, 0, 0};
    a = WMMA_BF16(ar0, loadB(wimg, 0  + nt, lane), a);
    a = WMMA_BF16(ar1, loadB(wimg, 4  + nt, lane), a);
    a = WMMA_BF16(ac0, loadB(wimg, 8  + nt, lane), a);
    a = WMMA_BF16(ac1, loadB(wimg, 12 + nt, lane), a);
    acc[nt] = a;
  }
  // + rel_dist * eW1[128,:] + eb1, relu, -> LDS bf16 (C layout -> row-major tile)
#pragma unroll
  for (int nt = 0; nt < 4; ++nt) {
    const int   n   = (lane & 15) + nt * 16;
    const float w1l = eW1[128 * 64 + n];
    const float b1  = eb1[n];
#pragma unroll
    for (int r = 0; r < 8; ++r) {
      const float rdM = __shfl(rd, r + moff);
      float v = acc[nt][r] + rdM * w1l + b1;
      v = v > 0.0f ? v : 0.0f;
      s_hid[wv][(r + moff) * 64 + n] = f2bf_bits(v);
    }
  }
  WAIT_LDS();

  // ---- edge MLP layer 2: edge_feat = hidden @ eW2 + eb2
  const unsigned short* hl = &s_hid[wv][0];
  v16bf a0 = loadA16(hl + m * 64 + k0);
  v16bf a1 = loadA16(hl + m * 64 + k1);
  v8f ef[4];
#pragma unroll
  for (int nt = 0; nt < 4; ++nt) {
    v8f a = {0, 0, 0, 0, 0, 0, 0, 0};
    a = WMMA_BF16(a0, loadB(wimg, 16 + nt, lane), a);
    a = WMMA_BF16(a1, loadB(wimg, 20 + nt, lane), a);
    ef[nt] = a;
  }
  // bias, scatter-add agg[row], and stash bf16 copy for the w-MLP
#pragma unroll
  for (int nt = 0; nt < 4; ++nt) {
    const int   n  = (lane & 15) + nt * 16;
    const float b2 = eb2[n];
#pragma unroll
    for (int r = 0; r < 8; ++r) {
      const int   M  = r + moff;
      const float v  = ef[nt][r] + b2;
      const int   eM = e0 + M;
      if (eM < E) {
        const int rowM = __shfl(row, M);
        atomicAdd(&agg[(size_t)rowM * 64 + n], v);
      }
      s_hid[wv][M * 64 + n] = f2bf_bits(v);
    }
  }
  WAIT_LDS();

  // ---- w-MLP layer 1: relu(edge_feat @ wW1 + wb1) -> LDS f32
  a0 = loadA16(hl + m * 64 + k0);
  a1 = loadA16(hl + m * 64 + k1);
#pragma unroll
  for (int nt = 0; nt < 4; ++nt) {
    v8f a = {0, 0, 0, 0, 0, 0, 0, 0};
    a = WMMA_BF16(a0, loadB(wimg, 24 + nt, lane), a);
    a = WMMA_BF16(a1, loadB(wimg, 28 + nt, lane), a);
    const int n  = (lane & 15) + nt * 16;
    const float b = wb1[n];
#pragma unroll
    for (int r = 0; r < 8; ++r) {
      float v = a[r] + b;
      s_hw[wv][(r + moff) * 64 + n] = v > 0.0f ? v : 0.0f;
    }
  }
  WAIT_LDS();

  // ---- w = hw . wW2 + wb2 ; scatter v_sum / cnt (lanes 0..15, one edge each)
  if (lane < 16 && (e0 + m) < E) {
    const float4* hw4 = (const float4*)(&s_hw[wv][m * 64]);
    const float4* w4  = (const float4*)wW2;
    float wacc = wb2[0];
#pragma unroll
    for (int k = 0; k < 16; ++k) {
      float4 q = hw4[k], w = w4[k];
      wacc += q.x * w.x + q.y * w.y + q.z * w.z + q.w * w.w;
    }
    atomicAdd(&vsum[(size_t)row * 3 + 0], s0 * wacc);
    atomicAdd(&vsum[(size_t)row * 3 + 1], s1 * wacc);
    atomicAdd(&vsum[(size_t)row * 3 + 2], s2 * wacc);
    atomicAdd(&cnt[row], 1.0f);
  }
}

// ---------------------------------------------------------------------------
// 4) fused node kernel: h_out = h_emb + relu([h_emb|agg]@nW1 + nb1)@nW2 + nb2
__global__ void __launch_bounds__(256)
egnn_node(const unsigned short* __restrict__ hbf, const float* __restrict__ hemb,
          const float* __restrict__ aggf, const float* __restrict__ nb1,
          const float* __restrict__ nb2, const unsigned short* __restrict__ wimg,
          float* __restrict__ out_h, int N) {
  __shared__ __align__(16) unsigned short s_hid[8][16 * 64];

  const int wv   = threadIdx.x >> 5;
  const int lane = threadIdx.x & 31;
  const int tile = blockIdx.x * 8 + wv;
  const int tiles = (N + 15) >> 4;
  if (tile >= tiles) return;                  // uniform per wave

  const int m    = lane & 15;
  const int moff = (lane >= 16) ? 8 : 0;
  const int k0   = moff, k1 = 32 + moff;

  int node = tile * 16 + m; if (node >= N) node = N - 1;

  const unsigned short* hr = hbf + (size_t)node * 64;
  const float*          ag = aggf + (size_t)node * 64;
  const v16bf a0 = loadA16(hr + k0);
  const v16bf a1 = loadA16(hr + k1);
  const v16bf a2 = cvtA16(ag + k0);
  const v16bf a3 = cvtA16(ag + k1);

#pragma unroll
  for (int nt = 0; nt < 4; ++nt) {
    v8f a = {0, 0, 0, 0, 0, 0, 0, 0};
    a = WMMA_BF16(a0, loadB(wimg, 32 + nt, lane), a);
    a = WMMA_BF16(a1, loadB(wimg, 36 + nt, lane), a);
    a = WMMA_BF16(a2, loadB(wimg, 40 + nt, lane), a);
    a = WMMA_BF16(a3, loadB(wimg, 44 + nt, lane), a);
    const int n  = (lane & 15) + nt * 16;
    const float b = nb1[n];
#pragma unroll
    for (int r = 0; r < 8; ++r) {
      float v = a[r] + b;
      v = v > 0.0f ? v : 0.0f;
      s_hid[wv][(r + moff) * 64 + n] = f2bf_bits(v);
    }
  }
  WAIT_LDS();

  const unsigned short* hl = &s_hid[wv][0];
  const v16bf b0 = loadA16(hl + m * 64 + k0);
  const v16bf b1 = loadA16(hl + m * 64 + k1);
#pragma unroll
  for (int nt = 0; nt < 4; ++nt) {
    v8f a = {0, 0, 0, 0, 0, 0, 0, 0};
    a = WMMA_BF16(b0, loadB(wimg, 48 + nt, lane), a);
    a = WMMA_BF16(b1, loadB(wimg, 52 + nt, lane), a);
    const int n  = (lane & 15) + nt * 16;
    const float b = nb2[n];
#pragma unroll
    for (int r = 0; r < 8; ++r) {
      const int nod = tile * 16 + r + moff;
      if (nod < N) {
        const size_t idx = (size_t)nod * 64 + n;
        out_h[idx] = hemb[idx] + a[r] + b;
      }
    }
  }
}

// ---------------------------------------------------------------------------
// 5) finalize v (mean over edges, normalize) and copy x into output.
__global__ void egnn_vfinal(const float* __restrict__ x, const float* __restrict__ vsum,
                            const float* __restrict__ cnt, float* __restrict__ out, int N) {
  int i = blockIdx.x * blockDim.x + threadIdx.x;
  if (i >= N) return;
  float c = cnt[i]; c = c > 1.0f ? c : 1.0f;
  float v0 = vsum[(size_t)i * 3 + 0] / c;
  float v1 = vsum[(size_t)i * 3 + 1] / c;
  float v2 = vsum[(size_t)i * 3 + 2] / c;
  float nrm = sqrtf(v0 * v0 + v1 * v1 + v2 * v2);
  float inv = 1.0f / fmaxf(nrm, 1e-12f);
  float* ox = out + (size_t)N * 64;
  float* ov = ox + (size_t)N * 3;
  ox[(size_t)i * 3 + 0] = x[(size_t)i * 3 + 0];
  ox[(size_t)i * 3 + 1] = x[(size_t)i * 3 + 1];
  ox[(size_t)i * 3 + 2] = x[(size_t)i * 3 + 2];
  ov[(size_t)i * 3 + 0] = v0 * inv;
  ov[(size_t)i * 3 + 1] = v1 * inv;
  ov[(size_t)i * 3 + 2] = v2 * inv;
}

// ---------------------------------------------------------------------------
extern "C" void kernel_launch(void* const* d_in, const int* in_sizes, int n_in,
                              void* d_out, int out_size, void* d_ws, size_t ws_size,
                              hipStream_t stream) {
  (void)n_in; (void)out_size; (void)ws_size;
  const float* h    = (const float*)d_in[0];
  const float* x    = (const float*)d_in[1];
  const int*   ei   = (const int*)  d_in[2];
  const float* embW = (const float*)d_in[3];
  const float* embB = (const float*)d_in[4];
  const float* eW1  = (const float*)d_in[5];
  const float* eb1  = (const float*)d_in[6];
  const float* eW2  = (const float*)d_in[7];
  const float* eb2  = (const float*)d_in[8];
  const float* nW1  = (const float*)d_in[9];
  const float* nb1  = (const float*)d_in[10];
  const float* nW2  = (const float*)d_in[11];
  const float* nb2  = (const float*)d_in[12];
  const float* wW1  = (const float*)d_in[13];
  const float* wb1  = (const float*)d_in[14];
  const float* wW2  = (const float*)d_in[15];
  const float* wb2  = (const float*)d_in[16];

  const int N = in_sizes[0] / 16;   // h is (N,16)
  const int E = in_sizes[2] / 2;    // edge_index is (2,E)

  char*  ws  = (char*)d_ws;
  size_t off = 0;
  float* agg  = (float*)(ws + off); off += (size_t)N * 64 * sizeof(float);
  float* vsum = (float*)(ws + off); off += (size_t)N * 3  * sizeof(float);
  float* cnt  = (float*)(ws + off); off += (size_t)N      * sizeof(float);
  off = (off + 255) & ~(size_t)255;
  float*          hemb = (float*)(ws + off);          off += (size_t)N * 64 * sizeof(float);
  unsigned short* hbf  = (unsigned short*)(ws + off); off += (size_t)N * 64 * sizeof(unsigned short);
  off = (off + 255) & ~(size_t)255;
  unsigned short* wimg = (unsigned short*)(ws + off); off += 56 * 512 * sizeof(unsigned short);

  const size_t nz = (size_t)N * 68;  // agg + vsum + cnt are contiguous
  egnn_zero<<<(unsigned)((nz + 255) / 256), 256, 0, stream>>>(agg, nz);
  egnn_prep<<<56, 512, 0, stream>>>(eW1, eW2, wW1, nW1, nW2, wimg);
  egnn_embed<<<(unsigned)(((size_t)N * 64 + 255) / 256), 256, 0, stream>>>(h, embW, embB, hemb, hbf, N);

  const int tilesE = (E + 15) / 16;
  egnn_edge<<<(tilesE + 7) / 8, 256, 0, stream>>>(ei, x, hbf, eW1, eb1, eb2, wb1, wW2, wb2,
                                                  wimg, agg, vsum, cnt, N, E);
  const int tilesN = (N + 15) / 16;
  egnn_node<<<(tilesN + 7) / 8, 256, 0, stream>>>(hbf, hemb, agg, nb1, nb2, wimg,
                                                  (float*)d_out, N);
  egnn_vfinal<<<(N + 255) / 256, 256, 0, stream>>>(x, vsum, cnt, (float*)d_out, N);
}
